// ProjTensorResNet_53807350284656
// MI455X (gfx1250) — matrix operand order repaired and checked by reading, not compiled, not run on hardware
//
#include <hip/hip_runtime.h>
#include <hip/hip_bf16.h>

// ---------------------------------------------------------------------------
// ProjTensorResNet on gfx1250: bf16 WMMA GEMM scan + Newton-Schulz polar
// ---------------------------------------------------------------------------

typedef __attribute__((ext_vector_type(16))) __bf16 v16bf;
typedef __attribute__((ext_vector_type(8)))  float  v8f;

#define DMODEL 512            // DIM*K_RANK
#define MROWS  (2048 * 6)     // 12288
#define MTILE  (MROWS / 16)   // 768 tiles along M
#define NQUAD  (DMODEL / 64)  // 8 quads of 4 N-tiles

__device__ __forceinline__ __bf16 to_bf16(float f) { return (__bf16)f; }

// Load one 16x32 bf16 B fragment: B[k][n] = W[n][k]; element i: k=(hi?16:0)+i
__device__ __forceinline__ v16bf load_bfrag(const float* __restrict__ bp)
{
  v16bf bfr;
  #pragma unroll
  for (int i = 0; i < 16; i += 2) {
    float2 v = *(const float2*)(bp + i);
    bfr[i]     = to_bf16(v.x);
    bfr[i + 1] = to_bf16(v.y);
  }
  return bfr;
}

// C = A @ W^T  [MODE==1: + bias, relu, residual(A)]
// Also writes the history slice outT[..., t] for rows whose slot < 3
// (slots 3..5 are overwritten by the projection kernel afterwards).
// A: (12288,512) f32 row-major, W: (512,512) f32 row-major, C: (12288,512)
// outT: (B,6,512,33) -> flat index (row*512+col)*33 + t
template <int MODE>
__global__ __launch_bounds__(256) void gemm_step_kernel(
    const float* __restrict__ A, const float* __restrict__ W,
    const float* __restrict__ bias,
    float* __restrict__ C, float* __restrict__ outT, int t)
{
  const int wave = blockIdx.x * (blockDim.x >> 5) + (threadIdx.x >> 5);
  const int lane = threadIdx.x & 31;
  const int tm = wave / NQUAD;       // 0..767 : 16-row strip of A
  const int tq = wave % NQUAD;       // 0..7   : 64-col strip of C
  const int hi = lane >> 4;          // 0: lanes 0-15, 1: lanes 16-31
  const int lo = lane & 15;

  const int m = tm * 16 + lo;        // A row for this lane
  const float* __restrict__ arow = A + (size_t)m * DMODEL;
  const float* __restrict__ brow0 = W + (size_t)(tq * 64 + lo) * DMODEL;

  v8f acc0 = {}, acc1 = {}, acc2 = {}, acc3 = {};
  #pragma unroll 2
  for (int kk = 0; kk < DMODEL; kk += 32) {
    // A 16x32 bf16 fragment: element i (i = 2v+half):
    //   k = (hi ? 16 : 0) + i
    v16bf afr;
    #pragma unroll
    for (int i = 0; i < 16; i += 2) {
      const int k = (hi ? 16 : 0) + i;
      float2 v = *(const float2*)(arow + kk + k);
      afr[i]     = to_bf16(v.x);
      afr[i + 1] = to_bf16(v.y);
    }
    // Stage all 4 B fragments in distinct buffers, then issue 4 WMMAs
    // back-to-back (no D->A/B RAW between them => no hazard NOPs needed,
    // and converts don't WAR the in-flight WMMA sources).
    const float* bbase = brow0 + kk + (hi ? 16 : 0);
    v16bf b0 = load_bfrag(bbase + (size_t)(0 * 16) * DMODEL);
    v16bf b1 = load_bfrag(bbase + (size_t)(1 * 16) * DMODEL);
    v16bf b2 = load_bfrag(bbase + (size_t)(2 * 16) * DMODEL);
    v16bf b3 = load_bfrag(bbase + (size_t)(3 * 16) * DMODEL);

    acc0 = __builtin_amdgcn_wmma_f32_16x16x32_bf16(false, afr, false, b0,
                                                   (short)0, acc0, false, false);
    acc1 = __builtin_amdgcn_wmma_f32_16x16x32_bf16(false, afr, false, b1,
                                                   (short)0, acc1, false, false);
    acc2 = __builtin_amdgcn_wmma_f32_16x16x32_bf16(false, afr, false, b2,
                                                   (short)0, acc2, false, false);
    acc3 = __builtin_amdgcn_wmma_f32_16x16x32_bf16(false, afr, false, b3,
                                                   (short)0, acc3, false, false);
  }

  // C/D layout: VGPR r -> M = r (+8 for upper half-lanes); lane -> N
  #pragma unroll
  for (int j = 0; j < 4; ++j) {
    const v8f& acc = (j == 0) ? acc0 : (j == 1) ? acc1 : (j == 2) ? acc2 : acc3;
    const int col = tq * 64 + j * 16 + lo;
    const float bval = (MODE == 1) ? bias[col] : 0.f;
    #pragma unroll
    for (int r = 0; r < 8; ++r) {
      const int row = tm * 16 + r + (hi ? 8 : 0);
      const size_t idx = (size_t)row * DMODEL + col;
      float v = acc[r];
      if (MODE == 1) {
        v = A[idx] + fmaxf(v + bval, 0.f);   // residual + relu(Wx+b)
      }
      C[idx] = v;
      if ((row % 6) < 3) outT[idx * 33 + t] = v;   // history for slots 0..2
    }
  }
}

// ---------------------------------------------------------------------------
// Polar projection of slots 3 (3x3, zero-padded) and 4,5 (32x16) in place,
// plus the history write for those slots.
// One wave (32 lanes) per (batch, slot). Newton-Schulz: A <- 1.5A - 0.5 A AtA
// ---------------------------------------------------------------------------
#define NS_ITERS 16

__global__ __launch_bounds__(32) void proj_slots_kernel(
    float* __restrict__ X, float* __restrict__ outT, int t)
{
  __shared__ float As[32 * 16];
  __shared__ float Ss[16 * 16];

  const int b     = blockIdx.x / 3;
  const int which = blockIdx.x % 3;       // 0 -> slot3, 1 -> slot4, 2 -> slot5
  const int slot  = 3 + which;
  const int lane  = threadIdx.x;
  const size_t base = ((size_t)b * 6 + slot) * DMODEL;
  float* Mp = X + base;
  float* Tp = outT + base * 33;           // + elem*33 + t

  if (which == 0) {
    // --- 3x3 polar, scalar (redundant across lanes), then zero-pad slot ---
    float a[3][3];
    #pragma unroll
    for (int p = 0; p < 3; ++p)
      #pragma unroll
      for (int q = 0; q < 3; ++q) a[p][q] = Mp[p * 16 + q];
    float fro = 0.f;
    #pragma unroll
    for (int p = 0; p < 3; ++p)
      #pragma unroll
      for (int q = 0; q < 3; ++q) fro += a[p][q] * a[p][q];
    const float inv = rsqrtf(fmaxf(fro, 1e-30f));
    #pragma unroll
    for (int p = 0; p < 3; ++p)
      #pragma unroll
      for (int q = 0; q < 3; ++q) a[p][q] *= inv;
    for (int it = 0; it < NS_ITERS; ++it) {
      float s[3][3];
      #pragma unroll
      for (int p = 0; p < 3; ++p)
        #pragma unroll
        for (int q = 0; q < 3; ++q) {
          float acc = 0.f;
          #pragma unroll
          for (int l = 0; l < 3; ++l) acc += a[l][p] * a[l][q];
          s[p][q] = acc;
        }
      float nx[3][3];
      #pragma unroll
      for (int p = 0; p < 3; ++p)
        #pragma unroll
        for (int q = 0; q < 3; ++q) {
          float acc = 0.f;
          #pragma unroll
          for (int l = 0; l < 3; ++l) acc += a[p][l] * s[l][q];
          nx[p][q] = 1.5f * a[p][q] - 0.5f * acc;
        }
      #pragma unroll
      for (int p = 0; p < 3; ++p)
        #pragma unroll
        for (int q = 0; q < 3; ++q) a[p][q] = nx[p][q];
    }
    // zero whole slot (jnp.pad), each lane owns one 16-element row
    #pragma unroll
    for (int q = 0; q < 16; ++q) {
      float v = 0.f;
      if (lane < 3 && q < 3) v = a[lane][q];
      Mp[lane * 16 + q] = v;
      Tp[(size_t)(lane * 16 + q) * 33 + t] = v;
    }
    return;
  }

  // --- 32x16 polar: lane owns one row of A ---
  float r[16];
  #pragma unroll
  for (int i = 0; i < 16; ++i) r[i] = Mp[lane * 16 + i];

  float fr = 0.f;
  #pragma unroll
  for (int i = 0; i < 16; ++i) fr += r[i] * r[i];
  #pragma unroll
  for (int off = 16; off > 0; off >>= 1) fr += __shfl_xor(fr, off, 32);
  const float inv = rsqrtf(fmaxf(fr, 1e-30f));
  #pragma unroll
  for (int i = 0; i < 16; ++i) r[i] *= inv;

  for (int it = 0; it < NS_ITERS; ++it) {
    #pragma unroll
    for (int i = 0; i < 16; ++i) As[lane * 16 + i] = r[i];
    __syncthreads();
    // S = A^T A : lane<16 -> (p=lane, q=0..7); lane>=16 -> (p=lane-16, q=8..15)
    const int p  = lane & 15;
    const int q0 = (lane >> 4) * 8;
    float sv[8];
    #pragma unroll
    for (int j = 0; j < 8; ++j) {
      float acc = 0.f;
      for (int l = 0; l < 32; ++l)
        acc += As[l * 16 + p] * As[l * 16 + q0 + j];
      sv[j] = acc;
    }
    #pragma unroll
    for (int j = 0; j < 8; ++j) Ss[p * 16 + q0 + j] = sv[j];
    __syncthreads();
    float nr[16];
    #pragma unroll
    for (int q = 0; q < 16; ++q) {
      float tt = 0.f;
      #pragma unroll
      for (int pp = 0; pp < 16; ++pp) tt += r[pp] * Ss[pp * 16 + q];
      nr[q] = 1.5f * r[q] - 0.5f * tt;
    }
    #pragma unroll
    for (int i = 0; i < 16; ++i) r[i] = nr[i];
  }

  #pragma unroll
  for (int i = 0; i < 16; ++i) {
    Mp[lane * 16 + i] = r[i];
    Tp[(size_t)(lane * 16 + i) * 33 + t] = r[i];
  }
}

// ---------------------------------------------------------------------------
// recon[b,c,i,j] = sum_{a,p,q} U1[c,a] G[a,p,q] U2[i,p] U3[j,q]  -> flat(B,3072)
// ---------------------------------------------------------------------------
__global__ __launch_bounds__(128) void recon_kernel(
    const float* __restrict__ Xlast, float* __restrict__ flat)
{
  __shared__ float G[3 * 16 * 16];
  __shared__ float U1s[9];
  __shared__ float U2s[32 * 16];
  __shared__ float U3s[32 * 16];
  __shared__ float T[3 * 32 * 16];

  const int b   = blockIdx.x;
  const int tid = threadIdx.x;
  const float* base = Xlast + (size_t)b * 6 * DMODEL;

  for (int i = tid; i < 768; i += 128) {
    const int a = i / 256, rem = i % 256, p = rem / 16, q = rem % 16;
    G[i] = base[a * DMODEL + p * 16 + q];        // slots 0..2, rows 0..15
  }
  for (int i = tid; i < 512; i += 128) {
    U2s[i] = base[4 * DMODEL + i];
    U3s[i] = base[5 * DMODEL + i];
  }
  if (tid < 9) U1s[tid] = base[3 * DMODEL + (tid / 3) * 16 + (tid % 3)];
  __syncthreads();

  // T[a][i][q] = sum_p G[a][p][q] * U2[i][p]
  for (int e = tid; e < 3 * 32 * 16; e += 128) {
    const int a = e / 512, rem = e % 512, i = rem / 16, q = rem % 16;
    float acc = 0.f;
    #pragma unroll
    for (int p = 0; p < 16; ++p) acc += G[a * 256 + p * 16 + q] * U2s[i * 16 + p];
    T[e] = acc;
  }
  __syncthreads();

  // recon[c][i][j] = sum_a U1[c,a] * sum_q T[a][i][q] * U3[j][q]
  for (int e = tid; e < 3 * 32 * 32; e += 128) {
    const int c = e / 1024, rem = e % 1024, i = rem / 32, j = rem % 32;
    float acc = 0.f;
    #pragma unroll
    for (int a = 0; a < 3; ++a) {
      float s = 0.f;
      #pragma unroll
      for (int q = 0; q < 16; ++q) s += T[a * 512 + i * 16 + q] * U3s[j * 16 + q];
      acc += U1s[c * 3 + a] * s;
    }
    flat[(size_t)b * 3072 + e] = acc;
  }
}

// ---------------------------------------------------------------------------
// logits = flat @ Wc^T + bc ; pred = softmax(logits)
// ---------------------------------------------------------------------------
__global__ __launch_bounds__(256) void classify_kernel(
    const float* __restrict__ flat, const float* __restrict__ Wc,
    const float* __restrict__ bc, float* __restrict__ pred,
    float* __restrict__ cls)
{
  __shared__ float red[256];
  __shared__ float logits[10];
  const int b = blockIdx.x, tid = threadIdx.x;
  const float* f = flat + (size_t)b * 3072;

  float acc[10];
  #pragma unroll
  for (int c = 0; c < 10; ++c) acc[c] = 0.f;
  for (int k = tid; k < 3072; k += 256) {
    const float x = f[k];
    #pragma unroll
    for (int c = 0; c < 10; ++c) acc[c] += x * Wc[c * 3072 + k];
  }
  for (int c = 0; c < 10; ++c) {
    red[tid] = acc[c];
    __syncthreads();
    for (int off = 128; off > 0; off >>= 1) {
      if (tid < off) red[tid] += red[tid + off];
      __syncthreads();
    }
    if (tid == 0) logits[c] = red[0] + bc[c];
    __syncthreads();
  }
  if (tid == 0) {
    float mx = -1e30f;
    #pragma unroll
    for (int c = 0; c < 10; ++c) mx = fmaxf(mx, logits[c]);
    float e[10], ssum = 0.f;
    #pragma unroll
    for (int c = 0; c < 10; ++c) { e[c] = expf(logits[c] - mx); ssum += e[c]; }
    const float invs = 1.f / ssum;
    #pragma unroll
    for (int c = 0; c < 10; ++c) {
      pred[(size_t)b * 10 + c] = e[c] * invs;
      cls[(size_t)b * 10 + c]  = logits[c];
    }
  }
}

// ---------------------------------------------------------------------------
extern "C" void kernel_launch(void* const* d_in, const int* in_sizes, int n_in,
                              void* d_out, int out_size, void* d_ws, size_t ws_size,
                              hipStream_t stream)
{
  const float* X  = (const float*)d_in[0];   // (2048, 6, 512)
  const float* W0 = (const float*)d_in[1];   // (512, 512)
  const float* W  = (const float*)d_in[2];   // (32, 512, 512)
  const float* bb = (const float*)d_in[3];   // (32, 512)
  const float* Wc = (const float*)d_in[4];   // (10, 3072)
  const float* bc = (const float*)d_in[5];   // (10,)
  float* out = (float*)d_out;

  const size_t MK = (size_t)MROWS * DMODEL;  // 6,291,456 floats
  float* Xa   = (float*)d_ws;
  float* Xb   = Xa + MK;
  float* flat = Xb + MK;                     // 2048*3072 floats

  float* outPred = out;                      // (2048,10)
  float* outCls  = out + (size_t)2048 * 10;  // (2048,10)
  float* outT    = out + (size_t)2048 * 10 * 2;  // (2048,6,512,33)

  const int gemmBlocks = (MTILE * NQUAD) / 8;   // 8 waves / block -> 768 blocks

  // step 0: Xc = project(X @ W0^T)
  gemm_step_kernel<0><<<gemmBlocks, 256, 0, stream>>>(X, W0, nullptr, Xa, outT, 0);
  proj_slots_kernel<<<2048 * 3, 32, 0, stream>>>(Xa, outT, 0);

  float* cur = Xa;
  float* nxt = Xb;
  for (int t = 0; t < 32; ++t) {
    gemm_step_kernel<1><<<gemmBlocks, 256, 0, stream>>>(
        cur, W + (size_t)t * DMODEL * DMODEL, bb + (size_t)t * DMODEL, nxt, outT, t + 1);
    proj_slots_kernel<<<2048 * 3, 32, 0, stream>>>(nxt, outT, t + 1);
    float* tmp = cur; cur = nxt; nxt = tmp;
  }

  recon_kernel<<<2048, 128, 0, stream>>>(cur, flat);
  classify_kernel<<<2048, 256, 0, stream>>>(flat, Wc, bc, outPred, outCls);
}
